// BasicTransformerBlock_62457414419273
// MI455X (gfx1250) — matrix operand order repaired and checked by reading, not compiled, not run on hardware
//
#include <hip/hip_runtime.h>
#include <hip/hip_bf16.h>

typedef __attribute__((ext_vector_type(16))) __bf16 bf16x16;
typedef __attribute__((ext_vector_type(8)))  __bf16 bf16x8;
typedef __attribute__((ext_vector_type(4)))  __bf16 bf16x4;
typedef __attribute__((ext_vector_type(8)))  float  floatx8;

#define DIM   2304
#define NHEAD 24
#define HDIM  96
#define SB    8192          // s*b = 2048*4 flattened rows
#define SPS   512           // sparse seq len
#define SPB   16            // sparse batch

__device__ __forceinline__ bf16x16 frag_merge(bf16x8 lo, bf16x8 hi) {
    return __builtin_shufflevector(lo, hi, 0, 1, 2, 3, 4, 5, 6, 7,
                                           8, 9, 10, 11, 12, 13, 14, 15);
}

// ---------------------------------------------------------------------------
// GEMM: C[M,N] = A[M,K] @ B[K,N] + bias[N]   (f32 in/out, bf16 WMMA compute)
// Block tile 128x128, K-step 32, double-buffered LDS with register prefetch.
// 8 waves in 4(M) x 2(N) grid; each wave owns 32x64 = 8 WMMA accumulators.
// All fragment reads are contiguous 16B runs -> ds_load_b128.
// ---------------------------------------------------------------------------
__global__ __launch_bounds__(256) void gemm_bias(
    const float* __restrict__ A, const float* __restrict__ B,
    const float* __restrict__ bias, float* __restrict__ C,
    int M, int N, int K)
{
    __shared__ alignas(16) __bf16 As[2][128][40];   // [buf][m][k], pitch 80B
    __shared__ alignas(16) __bf16 Bt[2][128][40];   // [buf][n][k], transposed

    const int tid  = threadIdx.x;
    const int lane = tid & 31;
    const int wave = tid >> 5;
    const int wm   = wave >> 1;          // 0..3 -> M offset wm*32
    const int wn   = wave & 1;           // 0..1 -> N offset wn*64
    const int half = lane >> 4;          // lane 0-15 vs 16-31
    const int l16  = lane & 15;
    const int m0   = blockIdx.y * 128;
    const int n0   = blockIdx.x * 128;

    floatx8 acc[2][4] = {};
    float4  pa[4], pb[4];

    // ---- prologue: fetch tile 0 into registers, store to LDS buffer 0
    #pragma unroll
    for (int u = 0; u < 4; ++u) {
        int it = tid + u * 256;
        pa[u] = *(const float4*)&A[(size_t)(m0 + (it >> 3)) * K + ((it & 7) << 2)];
        pb[u] = *(const float4*)&B[(size_t)(it >> 5) * N + n0 + ((it & 31) << 2)];
    }
    #pragma unroll
    for (int u = 0; u < 4; ++u) {
        int it = tid + u * 256;
        bf16x4 ha = { (__bf16)pa[u].x, (__bf16)pa[u].y, (__bf16)pa[u].z, (__bf16)pa[u].w };
        *(bf16x4*)&As[0][it >> 3][(it & 7) << 2] = ha;            // b64 store
        int r = it >> 5, c = (it & 31) << 2;
        Bt[0][c + 0][r] = (__bf16)pb[u].x;
        Bt[0][c + 1][r] = (__bf16)pb[u].y;
        Bt[0][c + 2][r] = (__bf16)pb[u].z;
        Bt[0][c + 3][r] = (__bf16)pb[u].w;
    }
    __syncthreads();

    const int nsteps = K >> 5;
    for (int s = 0; s < nsteps; ++s) {
        const int  cur  = s & 1;
        const bool more = (s + 1) < nsteps;

        // ---- prefetch next tile into registers (one load clause, no wait yet)
        if (more) {
            const int k0 = (s + 1) << 5;
            #pragma unroll
            for (int u = 0; u < 4; ++u) {
                int it = tid + u * 256;
                pa[u] = *(const float4*)&A[(size_t)(m0 + (it >> 3)) * K + k0 + ((it & 7) << 2)];
                pb[u] = *(const float4*)&B[(size_t)(k0 + (it >> 5)) * N + n0 + ((it & 31) << 2)];
            }
        }

        // ---- fragments from current LDS buffer (ds_load_b128 pairs)
        bf16x16 af[2], bfr[4];
        #pragma unroll
        for (int mi = 0; mi < 2; ++mi) {
            const int row = wm * 32 + mi * 16 + l16;
            bf16x8 lo = *(const bf16x8*)&As[cur][row][half * 8];
            bf16x8 hi = *(const bf16x8*)&As[cur][row][half * 8 + 16];
            af[mi] = frag_merge(lo, hi);
        }
        #pragma unroll
        for (int ni = 0; ni < 4; ++ni) {
            const int col = wn * 64 + ni * 16 + l16;
            bf16x8 lo = *(const bf16x8*)&Bt[cur][col][half * 8];
            bf16x8 hi = *(const bf16x8*)&Bt[cur][col][half * 8 + 16];
            bfr[ni] = frag_merge(lo, hi);
        }
        #pragma unroll
        for (int mi = 0; mi < 2; ++mi)
            #pragma unroll
            for (int ni = 0; ni < 4; ++ni)
                acc[mi][ni] = __builtin_amdgcn_wmma_f32_16x16x32_bf16(
                    false, af[mi], false, bfr[ni], (short)0, acc[mi][ni], false, false);

        // ---- drain prefetch registers into the other LDS buffer
        if (more) {
            const int nxt = cur ^ 1;
            #pragma unroll
            for (int u = 0; u < 4; ++u) {
                int it = tid + u * 256;
                bf16x4 ha = { (__bf16)pa[u].x, (__bf16)pa[u].y, (__bf16)pa[u].z, (__bf16)pa[u].w };
                *(bf16x4*)&As[nxt][it >> 3][(it & 7) << 2] = ha;
                int r = it >> 5, c = (it & 31) << 2;
                Bt[nxt][c + 0][r] = (__bf16)pb[u].x;
                Bt[nxt][c + 1][r] = (__bf16)pb[u].y;
                Bt[nxt][c + 2][r] = (__bf16)pb[u].z;
                Bt[nxt][c + 3][r] = (__bf16)pb[u].w;
            }
        }
        __syncthreads();
    }

    // ---- epilogue: D layout VGPR j, lane -> row = j + half*8, col = l16
    #pragma unroll
    for (int ni = 0; ni < 4; ++ni) {
        const int col = n0 + wn * 64 + ni * 16 + l16;
        const float bv = bias[col];
        #pragma unroll
        for (int mi = 0; mi < 2; ++mi) {
            #pragma unroll
            for (int j = 0; j < 8; ++j) {
                int row = m0 + wm * 32 + mi * 16 + j + half * 8;
                C[(size_t)row * N + col] = acc[mi][ni][j] + bv;
            }
        }
    }
}

// ---------------------------------------------------------------------------
// 3D RoPE on q and k in place. One thread per (row, head, pair); 48 pairs/head
// (3 axes x 16 freqs over 32-dim chunks, rotate-half pairing (i, i+16)).
// ---------------------------------------------------------------------------
__global__ void rope3d(float* __restrict__ q, float* __restrict__ k,
                       const int* __restrict__ fr, const int* __restrict__ hh,
                       const int* __restrict__ ww)
{
    size_t idx   = (size_t)blockIdx.x * blockDim.x + threadIdx.x;
    size_t total = (size_t)SB * NHEAD * 48;
    if (idx >= total) return;

    int   pair = (int)(idx % 48);
    int   head = (int)((idx / 48) % NHEAD);
    size_t row = idx / (48 * NHEAD);
    int   axis = pair >> 4;          // 0:t 1:h 2:w
    int   fi   = pair & 15;

    int s_i = (int)(row >> 2);       // b = 4
    int H = *hh, W = *ww; (void)fr;
    int pos;
    if      (axis == 0) pos = s_i / (H * W);
    else if (axis == 1) pos = (s_i / W) % H;
    else                pos = s_i % W;

    float inv_freq = __powf(10000.0f, -(float)(2 * fi) / 32.0f);
    float ang = (float)pos * inv_freq;
    float c = __cosf(ang), s = __sinf(ang);

    size_t base = row * DIM + (size_t)head * HDIM + axis * 32;
    float x1 = q[base + fi], x2 = q[base + 16 + fi];
    q[base + fi]      = x1 * c - x2 * s;
    q[base + 16 + fi] = x2 * c + x1 * s;
    x1 = k[base + fi]; x2 = k[base + 16 + fi];
    k[base + fi]      = x1 * c - x2 * s;
    k[base + 16 + fi] = x2 * c + x1 * s;
}

// ---------------------------------------------------------------------------
// Flash attention: 2 waves per block; each wave owns one 16-row q tile, both
// share the staged K/V chunk (each chunk feeds 24 WMMAs). K staged row-major
// (frag runs contiguous along d); V staged transposed (contiguous along key).
// ---------------------------------------------------------------------------
__device__ __forceinline__ float redmax16(float v) {
    #pragma unroll
    for (int m = 8; m >= 1; m >>= 1) v = fmaxf(v, __shfl_xor(v, m, 32));
    return v;
}
__device__ __forceinline__ float redsum16(float v) {
    #pragma unroll
    for (int m = 8; m >= 1; m >>= 1) v += __shfl_xor(v, m, 32);
    return v;
}

__global__ __launch_bounds__(64) void attn(
    const float* __restrict__ Q, const float* __restrict__ K,
    const float* __restrict__ V, float* __restrict__ O)
{
    __shared__ alignas(16) __bf16 Klds[32][104];    // [key][d], pitch 208B
    __shared__ alignas(16) __bf16 Vt[96][40];       // [d][key], pitch 80B
    __shared__ alignas(16) __bf16 Plds[2][16][32];  // per-wave probs tile

    const int tid  = threadIdx.x;
    const int wv_  = tid >> 5;         // wave 0/1
    const int lane = tid & 31;
    const int half = lane >> 4;
    const int l16  = lane & 15;
    const int qt   = blockIdx.x * 2 + wv_;   // 0..31 : q tile (one per wave)
    const int head = blockIdx.y;             // 0..23
    const int bb   = blockIdx.z;             // 0..15 : sparse batch
    const int hcol = head * HDIM;
    const float scale = 0.10206207f;         // 1/sqrt(96)

    // Q fragments (16x32 bf16, A layout), pre-scaled: 3 d-chunks of 32
    bf16x16 qf[3];
    {
        const size_t grow = (size_t)(qt * 16 + l16) * SPB + bb;  // sparse->flat row
        const float* qp = Q + grow * DIM + hcol;
        #pragma unroll
        for (int dc = 0; dc < 3; ++dc) {
            const int kb = half * 8;
            #pragma unroll
            for (int i = 0; i < 8; ++i) {
                qf[dc][i]     = (__bf16)(qp[dc * 32 + kb + i]      * scale);
                qf[dc][8 + i] = (__bf16)(qp[dc * 32 + kb + 16 + i] * scale);
            }
        }
    }

    floatx8 o[6] = {};
    float m_run[8], l_run[8];
    #pragma unroll
    for (int j = 0; j < 8; ++j) { m_run[j] = -1e30f; l_run[j] = 0.0f; }

    for (int c0 = 0; c0 < SPS; c0 += 32) {
        // ---- stage K/V chunk with all 64 threads: 2 threads per key row
        {
            const int krow_l = tid >> 1;            // 0..31
            const int part   = tid & 1;             // half-row: 12 float4 each
            const size_t krow = (size_t)(c0 + krow_l) * SPB + bb;
            const float4* kp = (const float4*)(K + krow * DIM + hcol) + part * 12;
            const float4* vp = (const float4*)(V + krow * DIM + hcol) + part * 12;
            const int d0 = part * 48;
            #pragma unroll
            for (int d4 = 0; d4 < 12; ++d4) {
                float4 f = kp[d4];
                bf16x4 h = { (__bf16)f.x, (__bf16)f.y, (__bf16)f.z, (__bf16)f.w };
                *(bf16x4*)&Klds[krow_l][d0 + d4 * 4] = h;       // b64 store
                f = vp[d4];
                Vt[d0 + d4 * 4 + 0][krow_l] = (__bf16)f.x;
                Vt[d0 + d4 * 4 + 1][krow_l] = (__bf16)f.y;
                Vt[d0 + d4 * 4 + 2][krow_l] = (__bf16)f.z;
                Vt[d0 + d4 * 4 + 3][krow_l] = (__bf16)f.w;
            }
        }
        __syncthreads();

        // ---- S = Q K^T for 2 key sub-tiles of 16
        floatx8 sc[2] = {};
        #pragma unroll
        for (int nt = 0; nt < 2; ++nt) {
            const int key = nt * 16 + l16;
            #pragma unroll
            for (int dc = 0; dc < 3; ++dc) {
                const int db = dc * 32 + half * 8;
                bf16x8 lo = *(const bf16x8*)&Klds[key][db];
                bf16x8 hi = *(const bf16x8*)&Klds[key][db + 16];
                sc[nt] = __builtin_amdgcn_wmma_f32_16x16x32_bf16(
                             false, qf[dc], false, frag_merge(lo, hi),
                             (short)0, sc[nt], false, false);
            }
        }

        // ---- online softmax update (row j lives in one 16-lane half)
        #pragma unroll
        for (int j = 0; j < 8; ++j) {
            float mx   = fmaxf(redmax16(sc[0][j]), redmax16(sc[1][j]));
            float mnew = fmaxf(m_run[j], mx);
            float alpha = __expf(m_run[j] - mnew);
            m_run[j] = mnew;
            float p0 = __expf(sc[0][j] - mnew);
            float p1 = __expf(sc[1][j] - mnew);
            l_run[j] = l_run[j] * alpha + redsum16(p0) + redsum16(p1);
            #pragma unroll
            for (int t = 0; t < 6; ++t) o[t][j] *= alpha;
            const int prow = j + half * 8;
            Plds[wv_][prow][l16]      = (__bf16)p0;
            Plds[wv_][prow][16 + l16] = (__bf16)p1;
        }

        // ---- P (16x32, A layout) @ V (32x96) -> accumulate O (wave-local)
        bf16x16 pf;
        {
            bf16x8 lo = *(const bf16x8*)&Plds[wv_][l16][half * 8];
            bf16x8 hi = *(const bf16x8*)&Plds[wv_][l16][half * 8 + 16];
            pf = frag_merge(lo, hi);
        }
        #pragma unroll
        for (int t = 0; t < 6; ++t) {
            const int vcol = t * 16 + l16;
            bf16x8 lo = *(const bf16x8*)&Vt[vcol][half * 8];
            bf16x8 hi = *(const bf16x8*)&Vt[vcol][half * 8 + 16];
            o[t] = __builtin_amdgcn_wmma_f32_16x16x32_bf16(
                       false, pf, false, frag_merge(lo, hi),
                       (short)0, o[t], false, false);
        }
        __syncthreads();   // protect K/V LDS before next chunk overwrite
    }

    // ---- normalize and store (same flat-row mapping as input)
    #pragma unroll
    for (int j = 0; j < 8; ++j) {
        const float inv_l = 1.0f / l_run[j];
        const size_t grow = (size_t)(qt * 16 + j + half * 8) * SPB + bb;
        #pragma unroll
        for (int t = 0; t < 6; ++t)
            O[grow * DIM + hcol + t * 16 + l16] = o[t][j] * inv_l;
    }
}

// ---------------------------------------------------------------------------
extern "C" void kernel_launch(void* const* d_in, const int* in_sizes, int n_in,
                              void* d_out, int out_size, void* d_ws, size_t ws_size,
                              hipStream_t stream)
{
    const float* hs = (const float*)d_in[0];
    const float* wq = (const float*)d_in[1];
    const float* bq = (const float*)d_in[2];
    const float* wk = (const float*)d_in[3];
    const float* bk = (const float*)d_in[4];
    const float* wv = (const float*)d_in[5];
    const float* bv = (const float*)d_in[6];
    const float* wo = (const float*)d_in[7];
    const float* bo = (const float*)d_in[8];
    const int*   fr = (const int*)d_in[9];
    const int*   hh = (const int*)d_in[10];
    const int*   ww = (const int*)d_in[11];

    const size_t matBytes = (size_t)SB * DIM * sizeof(float);
    float* Qb = (float*)d_ws;
    float* Kb = (float*)((char*)d_ws + 1 * matBytes);
    float* Vb = (float*)((char*)d_ws + 2 * matBytes);
    float* Ab = (float*)((char*)d_ws + 3 * matBytes);

    dim3 ggrid(DIM / 128, SB / 128), gblock(256);
    hipLaunchKernelGGL(gemm_bias, ggrid, gblock, 0, stream, hs, wq, bq, Qb, SB, DIM, DIM);
    hipLaunchKernelGGL(gemm_bias, ggrid, gblock, 0, stream, hs, wk, bk, Kb, SB, DIM, DIM);
    hipLaunchKernelGGL(gemm_bias, ggrid, gblock, 0, stream, hs, wv, bv, Vb, SB, DIM, DIM);

    size_t rope_total = (size_t)SB * NHEAD * 48;
    hipLaunchKernelGGL(rope3d, dim3((rope_total + 255) / 256), dim3(256), 0, stream,
                       Qb, Kb, fr, hh, ww);

    hipLaunchKernelGGL(attn, dim3(SPS / 32, NHEAD, SPB), dim3(64), 0, stream,
                       Qb, Kb, Vb, Ab);

    hipLaunchKernelGGL(gemm_bias, ggrid, gblock, 0, stream, Ab, wo, bo,
                       (float*)d_out, SB, DIM, DIM);
}
